// Attention_12515534700827
// MI455X (gfx1250) — matrix operand order repaired and checked by reading, not compiled, not run on hardware
//
#include <hip/hip_runtime.h>
#include <hip/hip_bf16.h>

typedef __attribute__((ext_vector_type(16))) _Float16 v16h;
typedef __attribute__((ext_vector_type(8)))  float    v8f;
typedef __attribute__((ext_vector_type(4)))  int      v4i;

#define N_HEADS 16
#define SEQ     2048
#define DIM     1024
#define HDIM    64
#define BATCH   2
#define MTOT    (BATCH * SEQ)          // 4096 rows
#define NEG_INF (-1.0e9f)

// -------- workspace layout (halves) --------
#define OFF_XH   ((size_t)0)
#define OFF_WQ   ((size_t)4194304)                 // 4096*1024
#define OFF_WK   (OFF_WQ + 1048576)
#define OFF_WV   (OFF_WK + 1048576)
#define OFF_WO   (OFF_WV + 1048576)
#define OFF_Q    (OFF_WO + 1048576)
#define OFF_K    (OFF_Q  + 4194304)
#define OFF_V    (OFF_K  + 4194304)
#define OFF_O    (OFF_V  + 4194304)

// -------- CDNA5 async global->LDS staging (guarded: compiles either way) ----
#if defined(__gfx1250__) && __has_builtin(__builtin_amdgcn_global_load_async_to_lds_b128) && __has_builtin(__builtin_amdgcn_s_wait_asynccnt)
#define HAVE_ASYNC_LDS 1
#else
#define HAVE_ASYNC_LDS 0
#endif

#define GLOBAL_AS __attribute__((address_space(1)))
#define LDS_AS    __attribute__((address_space(3)))

static __device__ __forceinline__ void stage16(_Float16* l, const _Float16* g) {
#if HAVE_ASYNC_LDS
    __builtin_amdgcn_global_load_async_to_lds_b128(
        (GLOBAL_AS v4i*)g, (LDS_AS v4i*)l, 0, 0);
#else
    *(uint4*)l = *(const uint4*)g;
#endif
}
static __device__ __forceinline__ void stage_wait() {
#if HAVE_ASYNC_LDS
    __builtin_amdgcn_s_wait_asynccnt(0);
#endif
}

static __device__ __forceinline__ v8f wmma_f16(v16h a, v16h b, v8f c) {
    return __builtin_amdgcn_wmma_f32_16x16x32_f16(false, a, false, b, (short)0, c, false, false);
}

// ---------------- fp32 -> f16 convert (vectorized x4) ----------------
__global__ __launch_bounds__(256) void cvt_f32_f16(const float* __restrict__ s,
                                                   _Float16* __restrict__ d, int n4) {
    int i = blockIdx.x * blockDim.x + threadIdx.x;
    if (i >= n4) return;
    float4 v = ((const float4*)s)[i];
    _Float16 h[4];
    h[0] = (_Float16)v.x; h[1] = (_Float16)v.y;
    h[2] = (_Float16)v.z; h[3] = (_Float16)v.w;
    *(uint2*)(d + 4 * (size_t)i) = *(const uint2*)h;
}

// ---------------- RoPE (in-place on Q and K, f16) ----------------
__global__ __launch_bounds__(256) void rope_kernel(_Float16* __restrict__ q,
                                                   _Float16* __restrict__ k,
                                                   const float* __restrict__ cosb,
                                                   const float* __restrict__ sinb) {
    int idx = blockIdx.x * blockDim.x + threadIdx.x;       // one (row, pair)
    if (idx >= MTOT * (DIM / 2)) return;
    int m = idx >> 9;                 // row 0..4095
    int p = idx & 511;                // pair index in row
    int h = p >> 5;                   // head
    int i = p & 31;                   // rotation index
    int s = m & (SEQ - 1);
    float c  = cosb[s * 32 + i];
    float sn = sinb[s * 32 + i];
    size_t off = (size_t)m * DIM + h * HDIM + 2 * i;
    float q0 = (float)q[off], q1 = (float)q[off + 1];
    q[off]     = (_Float16)(q0 * c - q1 * sn);
    q[off + 1] = (_Float16)(q0 * sn + q1 * c);
    float k0 = (float)k[off], k1 = (float)k[off + 1];
    k[off]     = (_Float16)(k0 * c - k1 * sn);
    k[off + 1] = (_Float16)(k0 * sn + k1 * c);
}

// ---------------- C = A @ W^T  (A:[M,K] f16, W:[N,K] f16) ----------------
// block = 256 threads (8 waves); tile 128(M) x 32(N); K stepped by 32.
// Double-buffered LDS: async-stage tile k+1 while computing tile k.
template <bool OUT_F32>
__global__ __launch_bounds__(256) void gemm_awt(const _Float16* __restrict__ A,
                                                const _Float16* __restrict__ W,
                                                _Float16* __restrict__ Ch,
                                                float* __restrict__ Cf,
                                                int M, int N, int K) {
    __shared__ _Float16 sA[2][128 * 32];      // [m][k]
    __shared__ _Float16 sWt[2][32 * 32];      // [n][k]
    const int t    = threadIdx.x;
    const int wave = t >> 5;
    const int lane = t & 31;
    const int hs   = lane >> 4;
    const int ln   = lane & 15;
    const int m_base = blockIdx.x * 128;
    const int n_base = blockIdx.y * 32;

    auto stage_tile = [&](int k0, int buf) {
        // A tile: 128x32 halves = 512 uint4; 2 per thread (async to LDS)
        #pragma unroll
        for (int i = 0; i < 2; ++i) {
            int idx = t + 256 * i;
            int row = idx >> 2, c8 = idx & 3;
            const _Float16* g = A + (size_t)(m_base + row) * K + k0 + c8 * 8;
            stage16(&sA[buf][row * 32 + c8 * 8], g);
            if (k0 + 64 < K) __builtin_prefetch(g + 64, 0, 1);
        }
        // W tile (32 n x 32 k), stored [n][k]
        if (t < 128) {
            int n = t >> 2, c8 = t & 3;
            const _Float16* src = W + (size_t)(n_base + n) * K + k0 + c8 * 8;
            #pragma unroll
            for (int j = 0; j < 8; ++j) sWt[buf][n * 32 + c8 * 8 + j] = src[j];
        }
    };

    v8f acc0 = {}, acc1 = {};
    stage_tile(0, 0);
    stage_wait();
    __syncthreads();
    for (int k0 = 0; k0 < K; k0 += 32) {
        const int buf = (k0 >> 5) & 1;
        if (k0 + 32 < K) stage_tile(k0 + 32, buf ^ 1);   // overlap DMA with compute

        v16h af, bf0, bf1;
        {   // A fragment: lane ln holds row, chunks k = 8*hs.. and 16+8*hs..
            const _Float16* ar = &sA[buf][(16 * wave + ln) * 32];
            ((uint4*)&af)[0] = *(const uint4*)(ar + 8 * hs);
            ((uint4*)&af)[1] = *(const uint4*)(ar + 16 + 8 * hs);
        }
        {   // B fragments: lane ln = n, elements = k chunks 8*hs.. and 16+8*hs.. (two 16-col halves)
            const uint4* q  = (const uint4*)&sWt[buf][ln * 32];
            const uint4* q2 = (const uint4*)&sWt[buf][(ln + 16) * 32];
            ((uint4*)&bf0)[0] = q[hs]; ((uint4*)&bf0)[1] = q[2 + hs];
            ((uint4*)&bf1)[0] = q2[hs]; ((uint4*)&bf1)[1] = q2[2 + hs];
        }
        acc0 = wmma_f16(af, bf0, acc0);
        acc1 = wmma_f16(af, bf1, acc1);

        stage_wait();          // this iteration's async writes (next buffer) done
        __syncthreads();       // publish next buffer to all waves
    }
    #pragma unroll
    for (int j = 0; j < 8; ++j) {
        int row = m_base + 16 * wave + j + 8 * hs;
        int col = n_base + ln;
        if (OUT_F32) {
            Cf[(size_t)row * N + col]      = acc0[j];
            Cf[(size_t)row * N + col + 16] = acc1[j];
        } else {
            Ch[(size_t)row * N + col]      = (_Float16)acc0[j];
            Ch[(size_t)row * N + col + 16] = (_Float16)acc1[j];
        }
    }
}

// ---------------- flash attention: per (b,h) head, 128 q-rows per block ----------------
__global__ __launch_bounds__(256) void flash_attn(const _Float16* __restrict__ qh,
                                                  const _Float16* __restrict__ kh,
                                                  const _Float16* __restrict__ vh,
                                                  _Float16* __restrict__ oh) {
    __shared__ _Float16 sKt[2][64 * 32];     // [kcol][d]
    __shared__ _Float16 sV[2][32 * 64];      // [kcol][d]
    __shared__ _Float16 sP[8][16][32];       // per-wave prob staging
    const int t    = threadIdx.x;
    const int wave = t >> 5;
    const int lane = t & 31;
    const int hs   = lane >> 4;
    const int ln   = lane & 15;
    const int bh = blockIdx.x;
    const int b  = bh >> 4;
    const int h  = bh & 15;
    const int q0 = blockIdx.y * 128;
    const int rowq = b * SEQ + q0 + 16 * wave;   // global row of this wave's M=0
    const int ktend = (q0 + 128) >> 5;           // causal extent in 32-col tiles

    auto stage_kv = [&](int kt, int buf) {
        // K (transposed, DS stores) + V (async) tiles: 32 rows x 64 d
        int row = t >> 3, c8 = t & 7;
        size_t g = (size_t)(b * SEQ + kt * 32 + row) * DIM + h * HDIM + c8 * 8;
        const _Float16* ks = kh + g;
        #pragma unroll
        for (int j2 = 0; j2 < 8; ++j2) sKt[buf][row * 64 + c8 * 8 + j2] = ks[j2];
        stage16(&sV[buf][row * 64 + c8 * 8], vh + g);
        if (kt + 2 < ktend) {
            __builtin_prefetch(ks + 64 * DIM, 0, 1);
            __builtin_prefetch(vh + g + 64 * DIM, 0, 1);
        }
    };

    // Q fragments (two K-dim steps of 32), pre-scaled by 1/sqrt(64)
    v16h qa0, qa1;
    {
        const _Float16* base = qh + (size_t)(rowq + ln) * DIM + h * HDIM;
        ((uint4*)&qa0)[0] = *(const uint4*)(base + 8 * hs);
        ((uint4*)&qa0)[1] = *(const uint4*)(base + 16 + 8 * hs);
        ((uint4*)&qa1)[0] = *(const uint4*)(base + 32 + 8 * hs);
        ((uint4*)&qa1)[1] = *(const uint4*)(base + 48 + 8 * hs);
        const _Float16 sc = (_Float16)0.125f;
        #pragma unroll
        for (int i = 0; i < 16; ++i) { qa0[i] *= sc; qa1[i] *= sc; }
    }

    float mrow[8], lrow[8];
    v8f oacc[4] = {};
    #pragma unroll
    for (int j = 0; j < 8; ++j) { mrow[j] = -3.0e38f; lrow[j] = 0.0f; }

    stage_kv(0, 0);
    stage_wait();
    __syncthreads();
    for (int kt = 0; kt < ktend; ++kt) {
        const int kbase = kt * 32;
        const int buf = kt & 1;
        if (kt + 1 < ktend) stage_kv(kt + 1, buf ^ 1);   // overlap DMA with compute

        // scores: 16 q-rows x 32 k-cols (two C fragments)
        v8f ca = {}, cb = {};
        #pragma unroll
        for (int ds = 0; ds < 2; ++ds) {
            v16h bf0, bf1;
            const uint4* p  = (const uint4*)&sKt[buf][ln * 64 + ds * 32];
            const uint4* p2 = (const uint4*)&sKt[buf][(ln + 16) * 64 + ds * 32];
            ((uint4*)&bf0)[0] = p[hs]; ((uint4*)&bf0)[1] = p[2 + hs];   // cols 0..15
            ((uint4*)&bf1)[0] = p2[hs]; ((uint4*)&bf1)[1] = p2[2 + hs]; // cols 16..31
            v16h af = ds ? qa1 : qa0;
            ca = wmma_f16(af, bf0, ca);
            cb = wmma_f16(af, bf1, cb);
        }
        // causal mask — only tiles straddling the diagonal (wave-uniform test,
        // EXEC stays all-ones so WMMA legality is unaffected)
        if (kbase + 31 > q0 + 16 * wave) {
            #pragma unroll
            for (int j = 0; j < 8; ++j) {
                int row  = q0 + 16 * wave + j + 8 * hs;
                int colA = kbase + ln;
                if (colA > row)      ca[j] += NEG_INF;
                if (colA + 16 > row) cb[j] += NEG_INF;
            }
        }
        // online softmax: row max across 16 lanes of each half-wave
        float mt[8];
        #pragma unroll
        for (int j = 0; j < 8; ++j) mt[j] = fmaxf(ca[j], cb[j]);
        #pragma unroll
        for (int off = 1; off < 16; off <<= 1)
            #pragma unroll
            for (int j = 0; j < 8; ++j) mt[j] = fmaxf(mt[j], __shfl_xor(mt[j], off, 32));
        float alpha[8];
        #pragma unroll
        for (int j = 0; j < 8; ++j) {
            float mn = fmaxf(mrow[j], mt[j]);
            alpha[j] = __expf(mrow[j] - mn);
            mrow[j]  = mn;
        }
        float pa[8], pb[8], ts[8];
        #pragma unroll
        for (int j = 0; j < 8; ++j) {
            pa[j] = __expf(ca[j] - mrow[j]);
            pb[j] = __expf(cb[j] - mrow[j]);
            ts[j] = pa[j] + pb[j];
        }
        #pragma unroll
        for (int off = 1; off < 16; off <<= 1)
            #pragma unroll
            for (int j = 0; j < 8; ++j) ts[j] += __shfl_xor(ts[j], off, 32);
        #pragma unroll
        for (int j = 0; j < 8; ++j) {
            lrow[j] = lrow[j] * alpha[j] + ts[j];
            oacc[0][j] *= alpha[j]; oacc[1][j] *= alpha[j];
            oacc[2][j] *= alpha[j]; oacc[3][j] *= alpha[j];
        }
        // C-layout probs -> A-layout fragment via per-wave LDS bounce
        #pragma unroll
        for (int j = 0; j < 8; ++j) {
            sP[wave][j + 8 * hs][ln]      = (_Float16)pa[j];
            sP[wave][j + 8 * hs][ln + 16] = (_Float16)pb[j];
        }
        v16h pf;
        {
            const _Float16* pr = &sP[wave][ln][0];
            ((uint4*)&pf)[0] = *(const uint4*)(pr + 8 * hs);
            ((uint4*)&pf)[1] = *(const uint4*)(pr + 16 + 8 * hs);
        }
        // O += P @ V  (4 d-chunks of 16)
        #pragma unroll
        for (int dc = 0; dc < 4; ++dc) {
            v16h vf;
            #pragma unroll
            for (int s = 0; s < 16; ++s) {
                int kk = (s < 8) ? (8 * hs + s) : (16 + 8 * hs + (s - 8));
                vf[s] = sV[buf][kk * 64 + dc * 16 + ln];
            }
            oacc[dc] = wmma_f16(pf, vf, oacc[dc]);
        }

        stage_wait();          // next tile's async writes complete
        __syncthreads();       // publish next buffer
    }
    // epilogue: normalize and store f16
    #pragma unroll
    for (int j = 0; j < 8; ++j) {
        float inv = 1.0f / lrow[j];
        int grow = rowq + j + 8 * hs;
        #pragma unroll
        for (int dc = 0; dc < 4; ++dc) {
            int col = h * HDIM + dc * 16 + ln;
            oh[(size_t)grow * DIM + col] = (_Float16)(oacc[dc][j] * inv);
        }
    }
}

extern "C" void kernel_launch(void* const* d_in, const int* in_sizes, int n_in,
                              void* d_out, int out_size, void* d_ws, size_t ws_size,
                              hipStream_t stream) {
    const float* x    = (const float*)d_in[0];
    const float* fcos = (const float*)d_in[1];
    const float* fsin = (const float*)d_in[2];
    // d_in[3] = mask (unused; causal mask applied analytically)
    const float* wq = (const float*)d_in[4];
    const float* wk = (const float*)d_in[5];
    const float* wv = (const float*)d_in[6];
    const float* wo = (const float*)d_in[7];
    float* out = (float*)d_out;

    _Float16* ws  = (_Float16*)d_ws;
    _Float16* xh  = ws + OFF_XH;
    _Float16* wqh = ws + OFF_WQ;
    _Float16* wkh = ws + OFF_WK;
    _Float16* wvh = ws + OFF_WV;
    _Float16* woh = ws + OFF_WO;
    _Float16* qh  = ws + OFF_Q;
    _Float16* kh  = ws + OFF_K;
    _Float16* vh  = ws + OFF_V;
    _Float16* oh  = ws + OFF_O;

    // 1) fp32 -> f16 casts
    cvt_f32_f16<<<(MTOT * DIM / 4 + 255) / 256, 256, 0, stream>>>(x, xh, MTOT * DIM / 4);
    cvt_f32_f16<<<(DIM * DIM / 4 + 255) / 256, 256, 0, stream>>>(wq, wqh, DIM * DIM / 4);
    cvt_f32_f16<<<(DIM * DIM / 4 + 255) / 256, 256, 0, stream>>>(wk, wkh, DIM * DIM / 4);
    cvt_f32_f16<<<(DIM * DIM / 4 + 255) / 256, 256, 0, stream>>>(wv, wvh, DIM * DIM / 4);
    cvt_f32_f16<<<(DIM * DIM / 4 + 255) / 256, 256, 0, stream>>>(wo, woh, DIM * DIM / 4);

    // 2) Q/K/V projections (WMMA)
    dim3 ggrid(MTOT / 128, DIM / 32);
    gemm_awt<false><<<ggrid, 256, 0, stream>>>(xh, wqh, qh, nullptr, MTOT, DIM, DIM);
    gemm_awt<false><<<ggrid, 256, 0, stream>>>(xh, wkh, kh, nullptr, MTOT, DIM, DIM);
    gemm_awt<false><<<ggrid, 256, 0, stream>>>(xh, wvh, vh, nullptr, MTOT, DIM, DIM);

    // 3) RoPE on Q and K
    rope_kernel<<<(MTOT * (DIM / 2) + 255) / 256, 256, 0, stream>>>(qh, kh, fcos, fsin);

    // 4) flash attention (WMMA, online softmax, causal)
    dim3 agrid(BATCH * N_HEADS, SEQ / 128);
    flash_attn<<<agrid, 256, 0, stream>>>(qh, kh, vh, oh);

    // 5) output projection -> fp32 result
    gemm_awt<true><<<ggrid, 256, 0, stream>>>(oh, woh, nullptr, out, MTOT, DIM, DIM);
}
